// CrossAttentionEncoder_52063593563037
// MI455X (gfx1250) — matrix operand order repaired and checked by reading, not compiled, not run on hardware
//
#include <hip/hip_runtime.h>
#include <hip/hip_bf16.h>

typedef __bf16 bf16_t;
typedef __attribute__((ext_vector_type(16))) __bf16 v16bf;
typedef __attribute__((ext_vector_type(8)))  __bf16 v8bf;
typedef __attribute__((ext_vector_type(8)))  float  v8f;
typedef __attribute__((address_space(3)))    bf16_t lds_bf16;

#define BQ     8
#define NPTS   4096
#define WIDTH  768
#define HEADS  12
#define DHEAD  64
#define NLAT   256
#define LAYERS 8
#define INDIM  54
#define MLAT   (BQ * NLAT)   // 2048
#define MPTS   (BQ * NPTS)   // 32768

enum { FLAG_GELU = 1, FLAG_F32OUT = 2, FLAG_RES = 4 };

__device__ __forceinline__ float gelu_exact(float x) {
    return 0.5f * x * (1.0f + erff(x * 0.70710678118654752f));
}

// issue one 16B async copy global -> LDS (ASYNCcnt); dst is LDS byte offset
__device__ __forceinline__ void async_copy_b128(const bf16_t* gsrc, void* lds_dst) {
    const unsigned loff = (unsigned)(unsigned long long)(lds_bf16*)lds_dst;
    asm volatile("global_load_async_to_lds_b128 %0, %1, off"
                 :: "v"(loff), "v"(gsrc) : "memory");
}
__device__ __forceinline__ void async_wait0() {
    asm volatile("s_wait_asynccnt 0" ::: "memory");
}

// ---------------------------------------------------------------------------
// Fourier embed + input projection: data = [pc, sin, cos, feats] @ in_W^T + b
// ---------------------------------------------------------------------------
__global__ __launch_bounds__(256)
void embed_in_kernel(const float* __restrict__ pc, const float* __restrict__ feats,
                     const float* __restrict__ inW, const float* __restrict__ inb,
                     bf16_t* __restrict__ D)
{
    __shared__ float emb[64];
    const int p   = blockIdx.x;
    const int tid = threadIdx.x;
    if (tid < 3) {
        emb[tid]      = pc[(size_t)p * 3 + tid];
        emb[51 + tid] = feats[(size_t)p * 3 + tid];
    }
    __syncthreads();
    if (tid < 24) {
        const int   c = tid >> 3, f = tid & 7;
        const float a = emb[c] * (float)(1 << f);
        emb[3 + tid]  = sinf(a);
        emb[27 + tid] = cosf(a);
    }
    __syncthreads();
#pragma unroll
    for (int oo = 0; oo < 3; ++oo) {
        const int    o  = tid + oo * 256;
        const float* wr = inW + (size_t)o * INDIM;
        float sum = inb[o];
#pragma unroll
        for (int k = 0; k < INDIM; ++k) sum += emb[k] * wr[k];
        D[(size_t)p * WIDTH + o] = (bf16_t)sum;
    }
}

__global__ __launch_bounds__(256)
void cvt_f32_bf16_kernel(const float* __restrict__ in, bf16_t* __restrict__ out, int n)
{
    const int i = blockIdx.x * 256 + threadIdx.x;
    if (i < n) out[i] = (bf16_t)in[i];
}

// ---------------------------------------------------------------------------
// LayerNorm: fp32 row (768) -> bf16 row
// ---------------------------------------------------------------------------
__global__ __launch_bounds__(256)
void ln_kernel(const float* __restrict__ X, const float* __restrict__ sc,
               const float* __restrict__ bi, bf16_t* __restrict__ H)
{
    const int row = blockIdx.x;
    const int tid = threadIdx.x;
    const float* x = X + (size_t)row * WIDTH;
    const float v0 = x[tid], v1 = x[tid + 256], v2 = x[tid + 512];
    float s  = v0 + v1 + v2;
    float sq = v0 * v0 + v1 * v1 + v2 * v2;
#pragma unroll
    for (int off = 16; off > 0; off >>= 1) {
        s  += __shfl_xor(s,  off, 32);
        sq += __shfl_xor(sq, off, 32);
    }
    __shared__ float ss[8], ssq[8];
    const int wave = tid >> 5, lane = tid & 31;
    if (lane == 0) { ss[wave] = s; ssq[wave] = sq; }
    __syncthreads();
    if (wave == 0) {
        float a  = (lane < 8) ? ss[lane]  : 0.0f;
        float b2 = (lane < 8) ? ssq[lane] : 0.0f;
#pragma unroll
        for (int off = 4; off > 0; off >>= 1) {
            a  += __shfl_xor(a,  off, 32);
            b2 += __shfl_xor(b2, off, 32);
        }
        if (lane == 0) { ss[0] = a; ssq[0] = b2; }
    }
    __syncthreads();
    const float mu   = ss[0] * (1.0f / WIDTH);
    const float var  = ssq[0] * (1.0f / WIDTH) - mu * mu;
    const float rstd = rsqrtf(var + 1e-5f);
    bf16_t* h = H + (size_t)row * WIDTH;
    h[tid]       = (bf16_t)((v0 - mu) * rstd * sc[tid]       + bi[tid]);
    h[tid + 256] = (bf16_t)((v1 - mu) * rstd * sc[tid + 256] + bi[tid + 256]);
    h[tid + 512] = (bf16_t)((v2 - mu) * rstd * sc[tid + 512] + bi[tid + 512]);
}

// ---------------------------------------------------------------------------
// WMMA GEMM: C[M,N] = A[M,K](bf16) @ W[N,K]^T(fp32->bf16) + bias, epilogue.
// Block: 256 threads = 8 waves (2 M x 4 N), block tile 64x128, K-step 32.
// Wave tile 32x32 = 4 x v_wmma_f32_16x16x32_bf16 per K-step.
// Double-buffered LDS; A tile staged with GLOBAL_LOAD_ASYNC_TO_LDS_B128,
// W tile staged through registers (fp32 -> bf16 convert).
// Requires M%64==0, N%128==0, K%32==0 (true for all GEMMs in this net).
// ---------------------------------------------------------------------------
__global__ __launch_bounds__(256, 2)
void gemm_bf16_kernel(const bf16_t* __restrict__ A, int lda,
                      const float* __restrict__ W, int ldw,
                      const float* __restrict__ bias,
                      void* __restrict__ Cout, int ldc,
                      const float* __restrict__ Res,
                      int K, int flags)
{
    __shared__ bf16_t As[2][64][40];
    __shared__ bf16_t Bs[2][128][40];
    const int tid  = threadIdx.x;
    const int wave = tid >> 5, lane = tid & 31;
    const int wr = wave & 1, wc = wave >> 1;          // 2 x 4 wave grid
    const int l15 = lane & 15, hi = lane >> 4;
    const int m0 = blockIdx.x * 64;
    const int n0 = blockIdx.y * 128;
    const int arow = tid >> 2, acg = (tid & 3) * 8;   // A stage: 64 x 32
    const int wrow = tid >> 1, wcg = (tid & 1) * 16;  // W stage: 128 x 32
    v8f acc00 = {}, acc01 = {}, acc10 = {}, acc11 = {};
    float wv[16];

    const int nk = K >> 5;

    // prologue: stage tile 0 into buffer 0
    async_copy_b128(A + (size_t)(m0 + arow) * lda + acg, &As[0][arow][acg]);
    {
        const float* ws = W + (size_t)(n0 + wrow) * ldw + wcg;
#pragma unroll
        for (int i = 0; i < 16; ++i) wv[i] = ws[i];
        v8bf lo, hi8;
#pragma unroll
        for (int i = 0; i < 8; ++i) { lo[i] = (bf16_t)wv[i]; hi8[i] = (bf16_t)wv[8 + i]; }
        *(v8bf*)&Bs[0][wrow][wcg]     = lo;
        *(v8bf*)&Bs[0][wrow][wcg + 8] = hi8;
    }
    async_wait0();
    __syncthreads();

    int cur = 0;
    for (int t = 0; t < nk; ++t) {
        const int  nxt = cur ^ 1;
        const bool have_next = (t + 1 < nk);
        const int  k1 = (t + 1) << 5;
        if (have_next) {
            async_copy_b128(A + (size_t)(m0 + arow) * lda + k1 + acg, &As[nxt][arow][acg]);
            const float* ws = W + (size_t)(n0 + wrow) * ldw + k1 + wcg;
#pragma unroll
            for (int i = 0; i < 16; ++i) wv[i] = ws[i];
        }

        // fragments from current buffer
        v16bf a0, a1, b0, b1;
        {
            const v8bf c0 = *(const v8bf*)&As[cur][wr * 32 + l15][hi * 8];
            const v8bf c1 = *(const v8bf*)&As[cur][wr * 32 + l15][16 + hi * 8];
            const v8bf c2 = *(const v8bf*)&As[cur][wr * 32 + 16 + l15][hi * 8];
            const v8bf c3 = *(const v8bf*)&As[cur][wr * 32 + 16 + l15][16 + hi * 8];
            const v8bf d0 = *(const v8bf*)&Bs[cur][wc * 32 + l15][hi * 8];
            const v8bf d1 = *(const v8bf*)&Bs[cur][wc * 32 + l15][16 + hi * 8];
            const v8bf d2 = *(const v8bf*)&Bs[cur][wc * 32 + 16 + l15][hi * 8];
            const v8bf d3 = *(const v8bf*)&Bs[cur][wc * 32 + 16 + l15][16 + hi * 8];
#pragma unroll
            for (int i = 0; i < 8; ++i) {
                a0[i] = c0[i]; a0[8 + i] = c1[i];
                a1[i] = c2[i]; a1[8 + i] = c3[i];
                b0[i] = d0[i]; b0[8 + i] = d1[i];
                b1[i] = d2[i]; b1[8 + i] = d3[i];
            }
        }
        acc00 = __builtin_amdgcn_wmma_f32_16x16x32_bf16(false, a0, false, b0, (short)0, acc00, false, false);
        acc01 = __builtin_amdgcn_wmma_f32_16x16x32_bf16(false, a0, false, b1, (short)0, acc01, false, false);
        acc10 = __builtin_amdgcn_wmma_f32_16x16x32_bf16(false, a1, false, b0, (short)0, acc10, false, false);
        acc11 = __builtin_amdgcn_wmma_f32_16x16x32_bf16(false, a1, false, b1, (short)0, acc11, false, false);

        if (have_next) {
            v8bf lo, hi8;
#pragma unroll
            for (int i = 0; i < 8; ++i) { lo[i] = (bf16_t)wv[i]; hi8[i] = (bf16_t)wv[8 + i]; }
            *(v8bf*)&Bs[nxt][wrow][wcg]     = lo;
            *(v8bf*)&Bs[nxt][wrow][wcg + 8] = hi8;
        }
        async_wait0();
        __syncthreads();
        cur = nxt;
    }

    // epilogue: C/D layout row = hi*8 + r, col = l15 per 16x16 tile
#pragma unroll
    for (int ar = 0; ar < 2; ++ar) {
        const int rowBase = m0 + wr * 32 + ar * 16 + hi * 8;
        const v8f accA = ar ? acc10 : acc00;
        const v8f accB = ar ? acc11 : acc01;
        const int colA = n0 + wc * 32 + l15;
        const int colB = colA + 16;
        const float biasA = bias ? bias[colA] : 0.0f;
        const float biasB = bias ? bias[colB] : 0.0f;
#pragma unroll
        for (int r = 0; r < 8; ++r) {
            const int row = rowBase + r;
            float vA = accA[r] + biasA;
            float vB = accB[r] + biasB;
            if (flags & FLAG_GELU) { vA = gelu_exact(vA); vB = gelu_exact(vB); }
            if (flags & FLAG_F32OUT) {
                float* o = (float*)Cout;
                if (flags & FLAG_RES) {
                    vA += Res[(size_t)row * ldc + colA];
                    vB += Res[(size_t)row * ldc + colB];
                }
                o[(size_t)row * ldc + colA] = vA;
                o[(size_t)row * ldc + colB] = vB;
            } else {
                bf16_t* o = (bf16_t*)Cout;
                o[(size_t)row * ldc + colA] = (bf16_t)vA;
                o[(size_t)row * ldc + colB] = (bf16_t)vB;
            }
        }
    }
}

// ---------------------------------------------------------------------------
// Flash attention (WMMA, online softmax). Block = 4 waves, 64 queries
// (16/wave), head dim 64 in registers, 32-key chunks staged in LDS
// (K via async-to-LDS copies, V transposed manually).
// ---------------------------------------------------------------------------
__global__ __launch_bounds__(128, 2)
void attn_kernel(const bf16_t* __restrict__ Q, long q_bs, int q_rs, int q_hs,
                 const bf16_t* __restrict__ Kp, const bf16_t* __restrict__ Vp,
                 long kv_bs, int kv_rs, int kv_hs,
                 int Lk, bf16_t* __restrict__ Out, long out_bs, int out_rs,
                 float sm_scale)
{
    __shared__ bf16_t Ks[32][72];        // [key][d]
    __shared__ bf16_t Vt[64][40];        // [d][key] (transposed)
    __shared__ bf16_t Pt[4][16][40];     // per-wave P transpose buffer
    const int tid  = threadIdx.x;
    const int wave = tid >> 5, lane = tid & 31;
    const int l15 = lane & 15, hi = lane >> 4;
    const int h = blockIdx.y, b = blockIdx.z;
    const int m0 = blockIdx.x * 64 + wave * 16;

    const bf16_t* qrow = Q + (size_t)b * q_bs + (size_t)(m0 + l15) * q_rs + (size_t)h * q_hs;
    v16bf qa0, qa1;
    {
        const v8bf c0 = *(const v8bf*)(qrow + hi * 8);
        const v8bf c1 = *(const v8bf*)(qrow + 16 + hi * 8);
        const v8bf c2 = *(const v8bf*)(qrow + 32 + hi * 8);
        const v8bf c3 = *(const v8bf*)(qrow + 48 + hi * 8);
#pragma unroll
        for (int i = 0; i < 8; ++i) {
            qa0[i] = c0[i]; qa0[8 + i] = c1[i];
            qa1[i] = c2[i]; qa1[8 + i] = c3[i];
        }
    }

    float mrow[8], lrow[8];
    v8f acc[4] = {};
#pragma unroll
    for (int r = 0; r < 8; ++r) { mrow[r] = -1e30f; lrow[r] = 0.0f; }

    const bf16_t* kb = Kp + (size_t)b * kv_bs + (size_t)h * kv_hs;
    const bf16_t* vb = Vp + (size_t)b * kv_bs + (size_t)h * kv_hs;
    const int skey = tid >> 2;           // 0..31
    const int sdg  = (tid & 3) * 16;     // 0,16,32,48

    for (int kc = 0; kc < Lk; kc += 32) {
        // stage K chunk via async copies, V chunk transposed via regs
        {
            const bf16_t* ks = kb + (size_t)(kc + skey) * kv_rs + sdg;
            async_copy_b128(ks,     &Ks[skey][sdg]);
            async_copy_b128(ks + 8, &Ks[skey][sdg + 8]);
            const bf16_t* vs = vb + (size_t)(kc + skey) * kv_rs + sdg;
#pragma unroll
            for (int i = 0; i < 16; ++i) Vt[sdg + i][skey] = vs[i];
        }
        async_wait0();
        __syncthreads();

        // scores: two 16x16 tiles (keys 0-15, 16-31), contraction d = 64
        v8f s0 = {}, s1 = {};
        {
            v16bf bk0, bk1, ck0, ck1;
            const v8bf p0 = *(const v8bf*)&Ks[l15][hi * 8];
            const v8bf p1 = *(const v8bf*)&Ks[l15][16 + hi * 8];
            const v8bf p2 = *(const v8bf*)&Ks[l15][32 + hi * 8];
            const v8bf p3 = *(const v8bf*)&Ks[l15][48 + hi * 8];
            const v8bf r0 = *(const v8bf*)&Ks[16 + l15][hi * 8];
            const v8bf r1 = *(const v8bf*)&Ks[16 + l15][16 + hi * 8];
            const v8bf r2 = *(const v8bf*)&Ks[16 + l15][32 + hi * 8];
            const v8bf r3 = *(const v8bf*)&Ks[16 + l15][48 + hi * 8];
#pragma unroll
            for (int i = 0; i < 8; ++i) {
                bk0[i] = p0[i]; bk0[8 + i] = p1[i];
                bk1[i] = p2[i]; bk1[8 + i] = p3[i];
                ck0[i] = r0[i]; ck0[8 + i] = r1[i];
                ck1[i] = r2[i]; ck1[8 + i] = r3[i];
            }
            s0 = __builtin_amdgcn_wmma_f32_16x16x32_bf16(false, qa0, false, bk0, (short)0, s0, false, false);
            s0 = __builtin_amdgcn_wmma_f32_16x16x32_bf16(false, qa1, false, bk1, (short)0, s0, false, false);
            s1 = __builtin_amdgcn_wmma_f32_16x16x32_bf16(false, qa0, false, ck0, (short)0, s1, false, false);
            s1 = __builtin_amdgcn_wmma_f32_16x16x32_bf16(false, qa1, false, ck1, (short)0, s1, false, false);
        }

        // online softmax (row = hi*8 + r, columns across the 16-lane half)
#pragma unroll
        for (int r = 0; r < 8; ++r) {
            const float a0 = s0[r] * sm_scale;
            const float a1 = s1[r] * sm_scale;
            float vmax = fmaxf(a0, a1);
            vmax = fmaxf(vmax, __shfl_xor(vmax, 1, 32));
            vmax = fmaxf(vmax, __shfl_xor(vmax, 2, 32));
            vmax = fmaxf(vmax, __shfl_xor(vmax, 4, 32));
            vmax = fmaxf(vmax, __shfl_xor(vmax, 8, 32));
            const float mnew = fmaxf(mrow[r], vmax);
            const float sf   = __expf(mrow[r] - mnew);
            mrow[r] = mnew;
            const float p0 = __expf(a0 - mnew);
            const float p1 = __expf(a1 - mnew);
            float ps = p0 + p1;
            ps += __shfl_xor(ps, 1, 32);
            ps += __shfl_xor(ps, 2, 32);
            ps += __shfl_xor(ps, 4, 32);
            ps += __shfl_xor(ps, 8, 32);
            lrow[r] = lrow[r] * sf + ps;
            acc[0][r] *= sf; acc[1][r] *= sf; acc[2][r] *= sf; acc[3][r] *= sf;
            Pt[wave][hi * 8 + r][l15]      = (bf16_t)p0;
            Pt[wave][hi * 8 + r][16 + l15] = (bf16_t)p1;
        }
        __syncthreads();

        // P as A-frag (16 x 32), V as B-frags from Vt
        v16bf pa;
        {
            const v8bf c0 = *(const v8bf*)&Pt[wave][l15][hi * 8];
            const v8bf c1 = *(const v8bf*)&Pt[wave][l15][16 + hi * 8];
#pragma unroll
            for (int i = 0; i < 8; ++i) { pa[i] = c0[i]; pa[8 + i] = c1[i]; }
        }
#pragma unroll
        for (int dt = 0; dt < 4; ++dt) {
            v16bf bv;
            const v8bf c0 = *(const v8bf*)&Vt[dt * 16 + l15][hi * 8];
            const v8bf c1 = *(const v8bf*)&Vt[dt * 16 + l15][16 + hi * 8];
#pragma unroll
            for (int i = 0; i < 8; ++i) { bv[i] = c0[i]; bv[8 + i] = c1[i]; }
            acc[dt] = __builtin_amdgcn_wmma_f32_16x16x32_bf16(false, pa, false, bv, (short)0, acc[dt], false, false);
        }
        __syncthreads();
    }

    bf16_t* ob = Out + (size_t)b * out_bs + (size_t)h * DHEAD;
#pragma unroll
    for (int dt = 0; dt < 4; ++dt) {
#pragma unroll
        for (int r = 0; r < 8; ++r) {
            const int row = m0 + hi * 8 + r;
            ob[(size_t)row * out_rs + dt * 16 + l15] = (bf16_t)(acc[dt][r] / lrow[r]);
        }
    }
}

// ---------------------------------------------------------------------------
// Host side
// ---------------------------------------------------------------------------
static void launch_gemm(const bf16_t* A, int lda, const float* W, int ldw,
                        const float* bias, void* C, int ldc, const float* Res,
                        int M, int N, int K, int flags, hipStream_t s)
{
    dim3 grid(M / 64, N / 128);
    gemm_bf16_kernel<<<grid, 256, 0, s>>>(A, lda, W, ldw, bias, C, ldc, Res, K, flags);
}

extern "C" void kernel_launch(void* const* d_in, const int* in_sizes, int n_in,
                              void* d_out, int out_size, void* d_ws, size_t ws_size,
                              hipStream_t stream)
{
    (void)in_sizes; (void)n_in; (void)out_size; (void)ws_size;
    const float* pc       = (const float*)d_in[0];
    const float* feats    = (const float*)d_in[1];
    const float* query    = (const float*)d_in[2];
    const float* in_W     = (const float*)d_in[3];
    const float* in_b     = (const float*)d_in[4];
    const float* ca_qkv_W = (const float*)d_in[5];
    const float* ca_qkv_b = (const float*)d_in[6];
    const float* ca_proj_W= (const float*)d_in[7];
    const float* ca_proj_b= (const float*)d_in[8];
    const float* ln1_s    = (const float*)d_in[9];
    const float* ln1_b    = (const float*)d_in[10];
    const float* qkv_W    = (const float*)d_in[11];
    const float* qkv_b    = (const float*)d_in[12];
    const float* proj_W   = (const float*)d_in[13];
    const float* proj_b   = (const float*)d_in[14];
    const float* ln2_s    = (const float*)d_in[15];
    const float* ln2_b    = (const float*)d_in[16];
    const float* fc_W     = (const float*)d_in[17];
    const float* fc_b     = (const float*)d_in[18];
    const float* fc2_W    = (const float*)d_in[19];
    const float* fc2_b    = (const float*)d_in[20];

    char* w = (char*)d_ws;
    auto carve = [&](size_t bytes) -> void* {
        void* p = (void*)w;
        w += (bytes + 255) & ~(size_t)255;
        return p;
    };
    bf16_t* D    = (bf16_t*)carve((size_t)MPTS * WIDTH * 2);
    bf16_t* KV   = (bf16_t*)carve((size_t)MPTS * 2 * WIDTH * 2);
    bf16_t* Qlat = (bf16_t*)carve((size_t)NLAT * WIDTH * 2);
    bf16_t* Qb   = (bf16_t*)carve((size_t)NLAT * WIDTH * 2);
    float*  X    = (float*) carve((size_t)MLAT * WIDTH * 4);
    bf16_t* Hb   = (bf16_t*)carve((size_t)MLAT * WIDTH * 2);
    bf16_t* QKV  = (bf16_t*)carve((size_t)MLAT * 3 * WIDTH * 2);
    bf16_t* Aout = (bf16_t*)carve((size_t)MLAT * WIDTH * 2);
    bf16_t* G    = (bf16_t*)carve((size_t)MLAT * 4 * WIDTH * 2);

    const float SM_SCALE = 0.125f;   // (d^-1/4)^2 with d = 64

    embed_in_kernel<<<MPTS, 256, 0, stream>>>(pc, feats, in_W, in_b, D);
    cvt_f32_bf16_kernel<<<(NLAT * WIDTH + 255) / 256, 256, 0, stream>>>(query, Qlat, NLAT * WIDTH);
    // kv = D @ ca_qkv_W[768:2304]^T + b
    launch_gemm(D, WIDTH, ca_qkv_W + (size_t)WIDTH * WIDTH, WIDTH,
                ca_qkv_b + WIDTH, KV, 2 * WIDTH, nullptr,
                MPTS, 2 * WIDTH, WIDTH, 0, stream);
    // q = query @ ca_qkv_W[0:768]^T + b (shared across batches)
    launch_gemm(Qlat, WIDTH, ca_qkv_W, WIDTH, ca_qkv_b, Qb, WIDTH, nullptr,
                NLAT, WIDTH, WIDTH, 0, stream);
    // cross attention: 256 latents attend to 4096 points
    attn_kernel<<<dim3(NLAT / 64, HEADS, BQ), 128, 0, stream>>>(
        Qb, 0L, WIDTH, DHEAD,
        KV, KV + WIDTH, (long)NPTS * 2 * WIDTH, 2 * WIDTH, DHEAD,
        NPTS, Aout, (long)NLAT * WIDTH, WIDTH, SM_SCALE);
    // latents = Aout @ ca_proj_W^T + b -> fp32 residual stream X
    launch_gemm(Aout, WIDTH, ca_proj_W, WIDTH, ca_proj_b, X, WIDTH, nullptr,
                MLAT, WIDTH, WIDTH, FLAG_F32OUT, stream);

    for (int l = 0; l < LAYERS; ++l) {
        ln_kernel<<<MLAT, 256, 0, stream>>>(X, ln1_s + (size_t)l * WIDTH,
                                            ln1_b + (size_t)l * WIDTH, Hb);
        launch_gemm(Hb, WIDTH, qkv_W + (size_t)l * 3 * WIDTH * WIDTH, WIDTH,
                    qkv_b + (size_t)l * 3 * WIDTH, QKV, 3 * WIDTH, nullptr,
                    MLAT, 3 * WIDTH, WIDTH, 0, stream);
        attn_kernel<<<dim3(NLAT / 64, HEADS, BQ), 128, 0, stream>>>(
            QKV,             (long)NLAT * 3 * WIDTH, 3 * WIDTH, 3 * DHEAD,
            QKV + DHEAD, QKV + 2 * DHEAD,
                             (long)NLAT * 3 * WIDTH, 3 * WIDTH, 3 * DHEAD,
            NLAT, Aout, (long)NLAT * WIDTH, WIDTH, SM_SCALE);
        launch_gemm(Aout, WIDTH, proj_W + (size_t)l * WIDTH * WIDTH, WIDTH,
                    proj_b + (size_t)l * WIDTH, X, WIDTH, X,
                    MLAT, WIDTH, WIDTH, FLAG_F32OUT | FLAG_RES, stream);
        ln_kernel<<<MLAT, 256, 0, stream>>>(X, ln2_s + (size_t)l * WIDTH,
                                            ln2_b + (size_t)l * WIDTH, Hb);
        launch_gemm(Hb, WIDTH, fc_W + (size_t)l * 4 * WIDTH * WIDTH, WIDTH,
                    fc_b + (size_t)l * 4 * WIDTH, G, 4 * WIDTH, nullptr,
                    MLAT, 4 * WIDTH, WIDTH, FLAG_GELU, stream);
        launch_gemm(G, 4 * WIDTH, fc2_W + (size_t)l * WIDTH * 4 * WIDTH, 4 * WIDTH,
                    fc2_b + (size_t)l * WIDTH, X, WIDTH, X,
                    MLAT, WIDTH, 4 * WIDTH, FLAG_F32OUT | FLAG_RES, stream);
    }

    float* out = (float*)d_out;
    hipMemcpyAsync(out, X, (size_t)MLAT * WIDTH * sizeof(float),
                   hipMemcpyDeviceToDevice, stream);
    hipMemcpyAsync(out + (size_t)MLAT * WIDTH, pc, (size_t)BQ * NPTS * 3 * sizeof(float),
                   hipMemcpyDeviceToDevice, stream);
}
